// myGNN_44942537786135
// MI455X (gfx1250) — matrix operand order repaired and checked by reading, not compiled, not run on hardware
//
#include <hip/hip_runtime.h>
#include <math.h>

// ---------------------------------------------------------------------------
// Problem constants (match reference)
// ---------------------------------------------------------------------------
#define BGRAPH 1024
#define NPG    128
#define DEGC   16
#define K1C    103            // ceil(0.8*128)
#define K2C    83             // ceil(0.8*103)
#define F_INC  30
#define H1C    30
#define H2C    50
#define FCC    100
#define NN1    (BGRAPH*NPG)   // 131072 nodes stage 1
#define NN2    (BGRAPH*K1C)   // 105472 nodes stage 2
#define NN3    (BGRAPH*K2C)   // 84992 nodes stage 3

typedef __attribute__((ext_vector_type(16))) _Float16 v16h;
typedef __attribute__((ext_vector_type(8)))  _Float16 v8h;
typedef __attribute__((ext_vector_type(8)))  float    v8f;

// ---------------------------------------------------------------------------
// Generic fill
// ---------------------------------------------------------------------------
__global__ void gnn_fill_f32(float* __restrict__ p, float v, int n) {
    int i = blockIdx.x * blockDim.x + threadIdx.x;
    if (i < n) p[i] = v;
}

// ---------------------------------------------------------------------------
// Pack A: f32 [M,K] row-major -> f16 [M,Kp] row-major, zero padded.
// ---------------------------------------------------------------------------
__global__ void gnn_pack_a(const float* __restrict__ A, _Float16* __restrict__ Ap,
                           int K, int Kp, int total /* = M*Kp */) {
    int i = blockIdx.x * blockDim.x + threadIdx.x;
    if (i >= total) return;
    int row = i / Kp;
    int k = i - row * Kp;
    float v = (k < K) ? A[(size_t)row * K + k] : 0.0f;
    Ap[i] = (_Float16)v;
}

// ---------------------------------------------------------------------------
// Pack B: f32 [K,N] row-major -> WMMA fragment order f16:
//   Bp[((kt*ntiles + nt)*32 + lane)*16 + e] = B[kt*32 + (e&7) + 8*(lane>>4) + 16*(e>>3)][nt*16 + (lane&15)]
// Each lane's 16 halves are contiguous (32B) and 16B aligned.
// ---------------------------------------------------------------------------
__global__ void gnn_pack_b(const float* __restrict__ Bm, _Float16* __restrict__ Bp,
                           int K, int N, int ntiles, int total /* = ktiles*ntiles*512 */) {
    int i = blockIdx.x * blockDim.x + threadIdx.x;
    if (i >= total) return;
    int e    = i & 15;
    int lane = (i >> 4) & 31;
    int tile = i >> 9;
    int nt = tile % ntiles;
    int kt = tile / ntiles;
    int lo = lane & 15, hi = lane >> 4;
    int k   = kt * 32 + (e & 7) + 8 * hi + 16 * (e >> 3);
    int col = nt * 16 + lo;
    float v = (k < K && col < N) ? Bm[(size_t)k * N + col] : 0.0f;
    Bp[i] = (_Float16)v;
}

// ---------------------------------------------------------------------------
// WMMA GEMM on pre-packed operands. One wave per 16x16 C tile; fully unrolled
// K loop (KP/32 steps of v_wmma_f32_16x16x32_f16). M % 16 == 0 at all call
// sites. blockDim.x == 256 (8 waves).
// ---------------------------------------------------------------------------
template<int KP>
__global__ void gnn_wmma_gemm(const _Float16* __restrict__ Ap,
                              const _Float16* __restrict__ Bp,
                              const float* __restrict__ bias,   // may be null
                              float* __restrict__ C,
                              int N, int ntiles_n, int total_tiles, int act)
{
    int wave = (int)((blockIdx.x * blockDim.x + threadIdx.x) >> 5);
    int lane = (int)(threadIdx.x & 31);
    if (wave >= total_tiles) return;          // wave-uniform; EXEC stays all-1s

    int mt = wave / ntiles_n;
    int nt = wave - mt * ntiles_n;
    int lo = lane & 15;
    int hi = lane >> 4;
    int row = (mt << 4) + lo;                 // A row handled by this lane
    int col = (nt << 4) + lo;                 // B/C column handled by this lane

    const _Float16* arow = Ap + (size_t)row * KP + 8 * hi;
    v8f acc = {};
#pragma unroll
    for (int kt = 0; kt < KP / 32; ++kt) {
        // A fragment: halves 0..7 -> K = kt*32 + 8*hi + i ; halves 8..15 -> +16
        v8h alo = *(const v8h*)(arow + kt * 32);
        v8h ahi = *(const v8h*)(arow + kt * 32 + 16);
        v16h a = __builtin_shufflevector(alo, ahi,
                                         0,1,2,3,4,5,6,7,8,9,10,11,12,13,14,15);
        // B fragment: pre-packed contiguous 32B per lane
        const _Float16* bptr = Bp + ((size_t)(kt * ntiles_n + nt) * 32 + lane) * 16;
        v8h blo = *(const v8h*)(bptr);
        v8h bhi = *(const v8h*)(bptr + 8);
        v16h b = __builtin_shufflevector(blo, bhi,
                                         0,1,2,3,4,5,6,7,8,9,10,11,12,13,14,15);
        acc = __builtin_amdgcn_wmma_f32_16x16x32_f16(
            /*neg_a=*/false, a, /*neg_b=*/false, b,
            /*c_mod=*/(short)0, acc, /*reuse_a=*/false, /*reuse_b=*/false);
    }

    if (col < N) {
        float bv = bias ? bias[col] : 0.0f;
#pragma unroll
        for (int r = 0; r < 8; ++r) {
            int rr = (mt << 4) + r + 8 * hi;  // C row for acc element r
            float v = acc[r] + bv;
            if (act == 1) v = fmaxf(v, 0.0f);
            C[(size_t)rr * N + col] = v;
        }
    }
}

// ---------------------------------------------------------------------------
// Degree accumulation: deg[dst] += 1 per active edge (deg pre-filled 1.0 for
// the self loop). `remap` null => stage-1 (all edges active).
// ---------------------------------------------------------------------------
__global__ void gnn_deg(const int* __restrict__ src, const int* __restrict__ dst,
                        const int* __restrict__ remap, float* __restrict__ deg, int E)
{
    int e = blockIdx.x * blockDim.x + threadIdx.x;
    if (e >= E) return;
    int s = src[e], d = dst[e];
    if (remap) {
        s = remap[s]; d = remap[d];
        if (s < 0 || d < 0) return;
    }
    atomicAdd(&deg[d], 1.0f);
}

// ---------------------------------------------------------------------------
// Edge scatter: agg[dst,:] += rsqrt(deg[src])*rsqrt(deg[dst]) * xw[src,:]
// ---------------------------------------------------------------------------
__global__ void gnn_scatter(const int* __restrict__ src, const int* __restrict__ dst,
                            const int* __restrict__ remap,
                            const float* __restrict__ xw, const float* __restrict__ deg,
                            float* __restrict__ agg, int F, int E)
{
    int e = blockIdx.x * blockDim.x + threadIdx.x;
    if (e >= E) return;
    int s = src[e], d = dst[e];
    if (remap) {
        s = remap[s]; d = remap[d];
        if (s < 0 || d < 0) return;
    }
    float norm = rsqrtf(deg[s]) * rsqrtf(deg[d]);
    const float* xr = xw + (size_t)s * F;
    float* ar = agg + (size_t)d * F;
    for (int c = 0; c < F; ++c) atomicAdd(&ar[c], norm * xr[c]);
}

// ---------------------------------------------------------------------------
// Finalize GCN conv (in place into agg): h = relu(agg + xw/deg + b)
// ---------------------------------------------------------------------------
__global__ void gnn_finalize(float* __restrict__ agg, const float* __restrict__ xw,
                             const float* __restrict__ deg, const float* __restrict__ bias,
                             int F, int total)
{
    int i = blockIdx.x * blockDim.x + threadIdx.x;
    if (i >= total) return;
    int node = i / F;
    int c = i - node * F;
    float v = agg[i] + xw[i] / deg[node] + bias[c];
    agg[i] = fmaxf(v, 0.0f);
}

// ---------------------------------------------------------------------------
// TopK score: s[i] = tanh( (h[i,:] . p) / ||p|| )
// ---------------------------------------------------------------------------
__global__ void gnn_score(const float* __restrict__ h, const float* __restrict__ p,
                          int F, int n, float* __restrict__ out)
{
    int i = blockIdx.x * blockDim.x + threadIdx.x;
    if (i >= n) return;
    float nrm = 0.0f, dot = 0.0f;
    const float* hr = h + (size_t)i * F;
    for (int c = 0; c < F; ++c) {
        float pv = p[c];
        nrm += pv * pv;
        dot += hr[c] * pv;
    }
    out[i] = tanhf(dot / sqrtf(nrm));
}

// ---------------------------------------------------------------------------
// Exact top-k per graph via O(n^2) rank-in-LDS. One block (128 thr) per graph.
// Reproduces descending sort with low-index tie-break (jax.lax.top_k order).
// ---------------------------------------------------------------------------
__global__ void gnn_topk(const float* __restrict__ scores, int n_in, int k_keep,
                         int* __restrict__ perm, float* __restrict__ vals,
                         int* __restrict__ remap)
{
    __shared__ float s[128];
    int g = blockIdx.x;
    int t = threadIdx.x;
    float my = -3.4e38f;
    if (t < n_in) my = scores[g * n_in + t];
    s[t] = my;
    __syncthreads();
    if (t >= n_in) return;
    int rank = 0;
    for (int j = 0; j < n_in; ++j) {
        float o = s[j];
        rank += (o > my) || ((o == my) && (j < t));
    }
    int node = g * n_in + t;
    if (rank < k_keep) {
        int outp = g * k_keep + rank;
        perm[outp] = node;
        vals[outp] = my;
        if (remap) remap[node] = outp;
    } else if (remap) {
        remap[node] = -1;
    }
}

// ---------------------------------------------------------------------------
// Gather + scale: out[j,:] = h[perm[j],:] * vals[j]
// ---------------------------------------------------------------------------
__global__ void gnn_gather(const float* __restrict__ h, const int* __restrict__ perm,
                           const float* __restrict__ vals, float* __restrict__ out,
                           int F, int total)
{
    int i = blockIdx.x * blockDim.x + threadIdx.x;
    if (i >= total) return;
    int j = i / F;
    int c = i - j * F;
    out[i] = h[(size_t)perm[j] * F + c] * vals[j];
}

// ---------------------------------------------------------------------------
// Global max + mean pool of kept stage-3 nodes. Block per graph, thread = ch.
// ---------------------------------------------------------------------------
__global__ void gnn_pool(const float* __restrict__ h, const int* __restrict__ perm,
                         const float* __restrict__ vals, int F, int k,
                         float* __restrict__ pooled)
{
    int g = blockIdx.x;
    int c = threadIdx.x;
    if (c >= F) return;
    float mx = -3.4e38f, sm = 0.0f;
    for (int j = 0; j < k; ++j) {
        int idx = perm[g * k + j];
        float v = h[(size_t)idx * F + c] * vals[g * k + j];
        mx = fmaxf(mx, v);
        sm += v;
    }
    pooled[g * (2 * F) + c] = mx;
    pooled[g * (2 * F) + F + c] = sm / (float)k;
}

// ---------------------------------------------------------------------------
// Final fc2 + sigmoid: out[b] = sigmoid(z[b,:] . W + b2)
// ---------------------------------------------------------------------------
__global__ void gnn_fc2(const float* __restrict__ z, const float* __restrict__ W,
                        const float* __restrict__ b2, float* __restrict__ out, int Bn, int F)
{
    int i = blockIdx.x * blockDim.x + threadIdx.x;
    if (i >= Bn) return;
    float acc = b2[0];
    const float* zr = z + (size_t)i * F;
    for (int c = 0; c < F; ++c) acc += zr[c] * W[c];
    out[i] = 1.0f / (1.0f + expf(-acc));
}

// ---------------------------------------------------------------------------
// Launch
// ---------------------------------------------------------------------------
static inline char* gnn_carve(char*& p, size_t bytes) {
    char* r = p;
    p += (bytes + 255) & ~(size_t)255;
    return r;
}

extern "C" void kernel_launch(void* const* d_in, const int* in_sizes, int n_in,
                              void* d_out, int out_size, void* d_ws, size_t ws_size,
                              hipStream_t stream) {
    const float* x       = (const float*)d_in[0];   // [NN1, 30]
    const int*   eidx    = (const int*)d_in[1];     // [2, E]
    // d_in[2] = batch (unused; graphs are equal-size)
    const float* conv1_W = (const float*)d_in[3];   // [30,30]
    const float* conv1_b = (const float*)d_in[4];   // [30]
    const float* pool1_p = (const float*)d_in[5];   // [30]
    const float* conv2_W = (const float*)d_in[6];   // [30,50]
    const float* conv2_b = (const float*)d_in[7];   // [50]
    const float* pool2_p = (const float*)d_in[8];   // [50]
    const float* fc1_W   = (const float*)d_in[9];   // [100,100]
    const float* fc1_b   = (const float*)d_in[10];  // [100]
    const float* fc2_W   = (const float*)d_in[11];  // [100,1]
    const float* fc2_b   = (const float*)d_in[12];  // [1]
    float* out = (float*)d_out;                     // [1024]

    const int E = in_sizes[1] / 2;
    const int* src = eidx;
    const int* dst = eidx + E;

    // --- workspace carve ---
    char* p = (char*)d_ws;
    float*    xw1     = (float*)   gnn_carve(p, (size_t)NN1 * F_INC * 4);
    float*    agg1    = (float*)   gnn_carve(p, (size_t)NN1 * H1C  * 4);  // becomes h1
    float*    deg1    = (float*)   gnn_carve(p, (size_t)NN1 * 4);
    float*    scores1 = (float*)   gnn_carve(p, (size_t)NN1 * 4);
    int*      perm1   = (int*)     gnn_carve(p, (size_t)NN2 * 4);
    float*    vals1   = (float*)   gnn_carve(p, (size_t)NN2 * 4);
    int*      remap1  = (int*)     gnn_carve(p, (size_t)NN1 * 4);
    float*    x2      = (float*)   gnn_carve(p, (size_t)NN2 * H1C * 4);
    float*    xw2     = (float*)   gnn_carve(p, (size_t)NN2 * H2C * 4);
    float*    agg2    = (float*)   gnn_carve(p, (size_t)NN2 * H2C * 4);   // becomes h2
    float*    deg2    = (float*)   gnn_carve(p, (size_t)NN2 * 4);
    float*    scores2 = (float*)   gnn_carve(p, (size_t)NN2 * 4);
    int*      perm2   = (int*)     gnn_carve(p, (size_t)NN3 * 4);
    float*    vals2   = (float*)   gnn_carve(p, (size_t)NN3 * 4);
    float*    pooled  = (float*)   gnn_carve(p, (size_t)BGRAPH * 2 * H2C * 4);
    float*    z       = (float*)   gnn_carve(p, (size_t)BGRAPH * FCC * 4);
    _Float16* xpad1   = (_Float16*)gnn_carve(p, (size_t)NN1 * 32 * 2);
    _Float16* x2pad   = (_Float16*)gnn_carve(p, (size_t)NN2 * 32 * 2);
    _Float16* ppad    = (_Float16*)gnn_carve(p, (size_t)BGRAPH * 128 * 2);
    _Float16* bpack1  = (_Float16*)gnn_carve(p, (size_t)1 * 2 * 512 * 2);  // kt=1, nt=2
    _Float16* bpack2  = (_Float16*)gnn_carve(p, (size_t)1 * 4 * 512 * 2);  // kt=1, nt=4
    _Float16* bpackfc = (_Float16*)gnn_carve(p, (size_t)4 * 7 * 512 * 2);  // kt=4, nt=7

    const int TB = 256;
#define GRID(n) (((n) + TB - 1) / TB)

    // --- init ---
    gnn_fill_f32<<<GRID(NN1 * H1C), TB, 0, stream>>>(agg1, 0.0f, NN1 * H1C);
    gnn_fill_f32<<<GRID(NN1), TB, 0, stream>>>(deg1, 1.0f, NN1);
    gnn_fill_f32<<<GRID(NN2 * H2C), TB, 0, stream>>>(agg2, 0.0f, NN2 * H2C);
    gnn_fill_f32<<<GRID(NN2), TB, 0, stream>>>(deg2, 1.0f, NN2);

    // --- conv1: xw1 = x @ conv1_W (WMMA, KP=32) ---
    {
        gnn_pack_a<<<GRID(NN1 * 32), TB, 0, stream>>>(x, xpad1, F_INC, 32, NN1 * 32);
        gnn_pack_b<<<GRID(1 * 2 * 512), TB, 0, stream>>>(conv1_W, bpack1, F_INC, H1C, 2, 1 * 2 * 512);
        int mt = NN1 / 16, nt = 2, total = mt * nt;
        gnn_wmma_gemm<32><<<(total + 7) / 8, TB, 0, stream>>>(
            xpad1, bpack1, nullptr, xw1, H1C, nt, total, 0);
    }
    gnn_deg<<<GRID(E), TB, 0, stream>>>(src, dst, nullptr, deg1, E);
    gnn_scatter<<<GRID(E), TB, 0, stream>>>(src, dst, nullptr, xw1, deg1, agg1, H1C, E);
    gnn_finalize<<<GRID(NN1 * H1C), TB, 0, stream>>>(agg1, xw1, deg1, conv1_b, H1C, NN1 * H1C);

    // --- pool1 ---
    gnn_score<<<GRID(NN1), TB, 0, stream>>>(agg1, pool1_p, H1C, NN1, scores1);
    gnn_topk<<<BGRAPH, 128, 0, stream>>>(scores1, NPG, K1C, perm1, vals1, remap1);
    gnn_gather<<<GRID(NN2 * H1C), TB, 0, stream>>>(agg1, perm1, vals1, x2, H1C, NN2 * H1C);

    // --- conv2: xw2 = x2 @ conv2_W (WMMA, KP=32) ---
    {
        gnn_pack_a<<<GRID(NN2 * 32), TB, 0, stream>>>(x2, x2pad, H1C, 32, NN2 * 32);
        gnn_pack_b<<<GRID(1 * 4 * 512), TB, 0, stream>>>(conv2_W, bpack2, H1C, H2C, 4, 1 * 4 * 512);
        int mt = NN2 / 16, nt = 4, total = mt * nt;
        gnn_wmma_gemm<32><<<(total + 7) / 8, TB, 0, stream>>>(
            x2pad, bpack2, nullptr, xw2, H2C, nt, total, 0);
    }
    gnn_deg<<<GRID(E), TB, 0, stream>>>(src, dst, remap1, deg2, E);
    gnn_scatter<<<GRID(E), TB, 0, stream>>>(src, dst, remap1, xw2, deg2, agg2, H2C, E);
    gnn_finalize<<<GRID(NN2 * H2C), TB, 0, stream>>>(agg2, xw2, deg2, conv2_b, H2C, NN2 * H2C);

    // --- pool2 ---
    gnn_score<<<GRID(NN2), TB, 0, stream>>>(agg2, pool2_p, H2C, NN2, scores2);
    gnn_topk<<<BGRAPH, 128, 0, stream>>>(scores2, K1C, K2C, perm2, vals2, nullptr);

    // --- readout: [max | mean] over kept nodes ---
    gnn_pool<<<BGRAPH, 64, 0, stream>>>(agg2, perm2, vals2, H2C, K2C, pooled);

    // --- fc1: z = relu(pooled @ fc1_W + fc1_b) (WMMA, KP=128, 4-wmma chain) ---
    {
        gnn_pack_a<<<GRID(BGRAPH * 128), TB, 0, stream>>>(pooled, ppad, 2 * H2C, 128, BGRAPH * 128);
        gnn_pack_b<<<GRID(4 * 7 * 512), TB, 0, stream>>>(fc1_W, bpackfc, FCC, FCC, 7, 4 * 7 * 512);
        int mt = BGRAPH / 16, nt = 7, total = mt * nt;
        gnn_wmma_gemm<128><<<(total + 7) / 8, TB, 0, stream>>>(
            ppad, bpackfc, fc1_b, z, FCC, nt, total, 1);
    }

    // --- fc2 + sigmoid ---
    gnn_fc2<<<GRID(BGRAPH), TB, 0, stream>>>(z, fc2_W, fc2_b, out, BGRAPH, FCC);

    (void)n_in; (void)out_size; (void)ws_size;
#undef GRID
}
